// Attention_46789373722975
// MI455X (gfx1250) — matrix operand order repaired and checked by reading, not compiled, not run on hardware
//
#include <hip/hip_runtime.h>

// Problem constants (match reference)
#define BQ     2
#define NSEQ   2048
#define DIMC   1024
#define NHEADS 16
#define HD     64
#define QKV_N  3072
#define MROWS  (BQ * NSEQ)   // 4096

typedef _Float16 h8  __attribute__((ext_vector_type(8)));
typedef _Float16 h16 __attribute__((ext_vector_type(16)));
typedef float    f8  __attribute__((ext_vector_type(8)));

static __device__ inline h16 cat8(h8 lo, h8 hi) {
  return __builtin_shufflevector(lo, hi, 0,1,2,3,4,5,6,7,8,9,10,11,12,13,14,15);
}

// A-matrix fragment (16x32 f16, M x K), A row-major [*, lda].
// Lane l<16 : row m0+l, elements 0..7 = K k0..k0+7, 8..15 = K k0+16..k0+23
// Lane l>=16: row m0+l-16, elements 0..7 = K k0+8..k0+15, 8..15 = K k0+24..k0+31
static __device__ inline h16 load_a_frag(const _Float16* A, int lda, int m0, int k0, int lane) {
  const int half = lane >> 4, r = lane & 15;
  const _Float16* p = A + (size_t)(m0 + r) * lda + k0 + half * 8;
  h8 lo = *(const h8*)(p);
  h8 hi = *(const h8*)(p + 16);
  return cat8(lo, hi);
}

// B-matrix fragment (32x16 f16, K x N) taken from an [N, K] row-major buffer
// (weights / K-contiguous), so each lane reads two runs of 8 contiguous f16.
// Lane l<16 : col n0+l, elements 0..7 = K k0..k0+7, 8..15 = K k0+16..k0+23
// Lane l>=16: col n0+l-16, elements 0..7 = K k0+8..k0+15, 8..15 = K k0+24..k0+31
static __device__ inline h16 load_b_frag(const _Float16* B, int ldb, int n0, int k0, int lane) {
  const int half = lane >> 4, r = lane & 15;
  const _Float16* p = B + (size_t)(n0 + r) * ldb + k0 + half * 8;
  h8 lo = *(const h8*)(p);
  h8 hi = *(const h8*)(p + 16);
  return cat8(lo, hi);
}

static __device__ inline f8 wmma32(h16 a, h16 b, f8 c) {
  return __builtin_amdgcn_wmma_f32_16x16x32_f16(false, a, false, b, (short)0, c, false, false);
}

// ---------------------------------------------------------------- elementwise
__global__ void f32_to_f16_kernel(const float* __restrict__ in, _Float16* __restrict__ out, int n) {
  int i = blockIdx.x * blockDim.x + threadIdx.x;
  if (i < n) out[i] = (_Float16)in[i];
}

// ---------------------------------------------------------------- WMMA GEMM
// C[M,N] = A[M,K](f16) * W[N,K]^T(f16) + bias[N]; C f32 row-major.
// Block = 4 waves; each wave computes a 32(M) x 64(N) tile; block tile 32 x 256.
// All 12 fragment loads of a k-step are issued before any WMMA so the loads
// clause together and the waits stay partial (overlap loads with WMMAs).
__global__ __launch_bounds__(128) void gemm_wmma_kernel(
    const _Float16* __restrict__ A, const _Float16* __restrict__ W,
    const float* __restrict__ bias, float* __restrict__ C,
    int M, int N, int K) {
  const int w = threadIdx.x >> 5, lane = threadIdx.x & 31;
  const int m0 = blockIdx.y * 32;
  const int n0 = blockIdx.x * 256 + w * 64;
  const int half = lane >> 4, r16 = lane & 15;

  f8 c00 = {}, c01 = {}, c02 = {}, c03 = {};   // rows m0..m0+15
  f8 c10 = {}, c11 = {}, c12 = {}, c13 = {};   // rows m0+16..m0+31
  for (int k0 = 0; k0 < K; k0 += 32) {
    // Issue every load of this k-step first (distinct live ranges -> one clause)
    h16 a0 = load_a_frag(A, K, m0,      k0, lane);
    h16 a1 = load_a_frag(A, K, m0 + 16, k0, lane);
    h16 b0 = load_b_frag(W, K, n0,      k0, lane);
    h16 b1 = load_b_frag(W, K, n0 + 16, k0, lane);
    h16 b2 = load_b_frag(W, K, n0 + 32, k0, lane);
    h16 b3 = load_b_frag(W, K, n0 + 48, k0, lane);
    if (k0 + 32 < K) {
      // GL2 prefetch of the next k-tile stream
      __builtin_prefetch(A + (size_t)(m0 + r16) * K + k0 + 32, 0, 3);
      __builtin_prefetch(W + (size_t)(n0 + r16) * K + k0 + 32, 0, 3);
    }
    c00 = wmma32(a0, b0, c00);
    c10 = wmma32(a1, b0, c10);
    c01 = wmma32(a0, b1, c01);
    c11 = wmma32(a1, b1, c11);
    c02 = wmma32(a0, b2, c02);
    c12 = wmma32(a1, b2, c12);
    c03 = wmma32(a0, b3, c03);
    c13 = wmma32(a1, b3, c13);
  }

  f8 cr0[4] = {c00, c01, c02, c03};
  f8 cr1[4] = {c10, c11, c12, c13};
#pragma unroll
  for (int j = 0; j < 4; ++j) {
    const int cn = n0 + j * 16 + r16;
    const float bv = bias ? bias[cn] : 0.0f;
#pragma unroll
    for (int r = 0; r < 8; ++r) {
      int row0 = m0 + r + 8 * half;
      C[(size_t)row0 * N + cn]        = cr0[j][r] + bv;
      C[(size_t)(row0 + 16) * N + cn] = cr1[j][r] + bv;
    }
  }
}

// ---------------------------------------------------------------- RoPE + pack
// qkv: [B,N,3*1024] f32.  Produces:
//   qh [B,H,N,64] f16 (rope, +q_bias, * HD^-0.5 folded in)
//   kh [B,H,N,64] f16 (rope)
//   vt [B,H,64,N] f16 (+v_bias, transposed for PV B-fragments)
__global__ void rope_pack_kernel(const float* __restrict__ qkv, const float* __restrict__ pe,
                                 const float* __restrict__ q_bias, const float* __restrict__ v_bias,
                                 _Float16* __restrict__ qh, _Float16* __restrict__ kh,
                                 _Float16* __restrict__ vt) {
  int idx = blockIdx.x * blockDim.x + threadIdx.x;       // [0, B*H*N*32)
  int d2 = idx & 31;
  int n  = (idx >> 5) & (NSEQ - 1);
  int h  = (idx >> 16) & (NHEADS - 1);
  int b  = idx >> 20;
  if (b >= BQ) return;

  size_t base = ((size_t)b * NSEQ + n) * (size_t)QKV_N;
  int col = h * HD + 2 * d2;

  float p00 = pe[(((size_t)n * 32 + d2) * 2 + 0) * 2 + 0];
  float p01 = pe[(((size_t)n * 32 + d2) * 2 + 0) * 2 + 1];
  float p10 = pe[(((size_t)n * 32 + d2) * 2 + 1) * 2 + 0];
  float p11 = pe[(((size_t)n * 32 + d2) * 2 + 1) * 2 + 1];

  float q0 = qkv[base + col]     + q_bias[col];
  float q1 = qkv[base + col + 1] + q_bias[col + 1];
  float k0 = qkv[base + DIMC + col];
  float k1 = qkv[base + DIMC + col + 1];
  float v0 = qkv[base + 2 * DIMC + col]     + v_bias[col];
  float v1 = qkv[base + 2 * DIMC + col + 1] + v_bias[col + 1];

  const float scale = 0.125f;                            // HD^-0.5 = 1/8
  float qo0 = p00 * q0 + p01 * q1, qo1 = p10 * q0 + p11 * q1;
  float ko0 = p00 * k0 + p01 * k1, ko1 = p10 * k0 + p11 * k1;

  size_t ob = (((size_t)(b * NHEADS + h)) * NSEQ + n) * HD + 2 * d2;
  qh[ob]     = (_Float16)(qo0 * scale);
  qh[ob + 1] = (_Float16)(qo1 * scale);
  kh[ob]     = (_Float16)ko0;
  kh[ob + 1] = (_Float16)ko1;

  size_t vb = (((size_t)(b * NHEADS + h)) * HD + 2 * d2) * NSEQ + n;
  vt[vb]        = (_Float16)v0;
  vt[vb + NSEQ] = (_Float16)v1;
}

// ---------------------------------------------------------------- k_mean
__global__ void kmean_kernel(const _Float16* __restrict__ kh, float* __restrict__ km) {
  int idx = blockIdx.x * blockDim.x + threadIdx.x;       // [0, B*N*64)
  int d = idx & (HD - 1);
  int n = (idx >> 6) & (NSEQ - 1);
  int b = idx >> 17;
  if (b >= BQ) return;
  float s = 0.0f;
#pragma unroll
  for (int h = 0; h < NHEADS; ++h)
    s += (float)kh[(((size_t)(b * NHEADS + h)) * NSEQ + n) * HD + d];
  km[idx] = s * (1.0f / NHEADS);
}

// ---------------------------------------------------------------- attention
// One wave handles 16 query rows; 4 waves per block; flash-style online softmax.
__global__ __launch_bounds__(128) void attn_kernel(const _Float16* __restrict__ qh,
                                                   const _Float16* __restrict__ kh,
                                                   const _Float16* __restrict__ vt,
                                                   float* __restrict__ attn_out) {
  __shared__ __align__(16) float    Sld[4][16][32];
  __shared__ __align__(16) _Float16 Pld[4][16][32];
  __shared__ float rowstat[4][16];

  const int w = threadIdx.x >> 5, lane = threadIdx.x & 31;
  const int half = lane >> 4, r16 = lane & 15;
  const int qtile = blockIdx.x * 4 + w;
  const int h = blockIdx.y, b = blockIdx.z;
  const int q0 = qtile * 16;

  const _Float16* Q = qh + ((size_t)(b * NHEADS + h)) * NSEQ * HD;
  const _Float16* K = kh + ((size_t)(b * NHEADS + h)) * NSEQ * HD;
  const _Float16* V = vt + ((size_t)(b * NHEADS + h)) * HD * NSEQ;

  // Q fragments for the wave's 16 rows, full K=64 contraction (2 A-frags)
  h16 aq0 = load_a_frag(Q, HD, q0, 0, lane);
  h16 aq1 = load_a_frag(Q, HD, q0, 32, lane);

  f8 o0 = {}, o1 = {}, o2 = {}, o3 = {};
  float m_old = -3.0e38f, l_old = 0.0f;   // owned per-row by lanes 0..15

  for (int kc = 0; kc < NSEQ; kc += 32) {
    // S tile: 16 queries x 32 keys (two 16x16 f32 fragments)
    h16 bk00 = load_b_frag(K, HD, kc,      0,  lane);
    h16 bk01 = load_b_frag(K, HD, kc,      32, lane);
    h16 bk10 = load_b_frag(K, HD, kc + 16, 0,  lane);
    h16 bk11 = load_b_frag(K, HD, kc + 16, 32, lane);
    f8 s0 = {}, s1 = {};
    s0 = wmma32(aq0, bk00, s0);
    s0 = wmma32(aq1, bk01, s0);
    s1 = wmma32(aq0, bk10, s1);
    s1 = wmma32(aq1, bk11, s1);

    // C-frag layout -> LDS row-major [row][key]
#pragma unroll
    for (int r = 0; r < 8; ++r) {
      Sld[w][r + 8 * half][r16]      = s0[r];
      Sld[w][r + 8 * half][r16 + 16] = s1[r];
    }
    asm volatile("s_wait_dscnt 0" ::: "memory");

    if (lane < 16) {                        // lane == row owner
      int row = lane;
      float rm = m_old;
      for (int j = 0; j < 32; ++j) rm = fmaxf(rm, Sld[w][row][j]);
      float corr = __expf(m_old - rm);
      float ls = 0.0f;
      for (int j = 0; j < 32; ++j) {
        float p = __expf(Sld[w][row][j] - rm);
        Pld[w][row][j] = (_Float16)p;
        ls += p;
      }
      l_old = l_old * corr + ls;
      m_old = rm;
      rowstat[w][row] = corr;
    }
    asm volatile("s_wait_dscnt 0" ::: "memory");

    // Rescale running output by per-row correction
#pragma unroll
    for (int r = 0; r < 8; ++r) {
      float cr = rowstat[w][r + 8 * half];
      o0[r] *= cr; o1[r] *= cr; o2[r] *= cr; o3[r] *= cr;
    }

    // P as an A-fragment straight from LDS (matches 16-bit A 16x32 layout)
    h16 ap;
    {
      const _Float16* pp = &Pld[w][r16][half * 8];
      h8 lo = *(const h8*)(pp);
      h8 hi = *(const h8*)(pp + 16);
      ap = cat8(lo, hi);
    }
    // O += P(16x32) x V(32x64): B-frags from transposed V (K-contiguous)
    h16 bv0 = load_b_frag(V, NSEQ, 0,  kc, lane);
    h16 bv1 = load_b_frag(V, NSEQ, 16, kc, lane);
    h16 bv2 = load_b_frag(V, NSEQ, 32, kc, lane);
    h16 bv3 = load_b_frag(V, NSEQ, 48, kc, lane);
    o0 = wmma32(ap, bv0, o0);
    o1 = wmma32(ap, bv1, o1);
    o2 = wmma32(ap, bv2, o2);
    o3 = wmma32(ap, bv3, o3);
  }

  if (lane < 16) rowstat[w][lane] = 1.0f / l_old;
  asm volatile("s_wait_dscnt 0" ::: "memory");

  float* O = attn_out + (((size_t)(b * NHEADS + h)) * NSEQ + q0) * HD;
#pragma unroll
  for (int r = 0; r < 8; ++r) {
    int row = r + 8 * half;
    float inv = rowstat[w][row];
    O[(size_t)row * HD + 0  + r16] = o0[r] * inv;
    O[(size_t)row * HD + 16 + r16] = o1[r] * inv;
    O[(size_t)row * HD + 32 + r16] = o2[r] * inv;
    O[(size_t)row * HD + 48 + r16] = o3[r] * inv;
  }
}

// ---------------------------------------------------------------- layernorm
// In: attn_out [B,H,N,64] f32; Out: yh [B*N, 1024] f16 normalized (gamma/beta).
__global__ __launch_bounds__(256) void ln_kernel(const float* __restrict__ attn_out,
                                                 const float* __restrict__ gamma,
                                                 const float* __restrict__ beta,
                                                 _Float16* __restrict__ yh) {
  __shared__ float red[256];
  const int row = blockIdx.x;            // b*N + n
  const int b = row / NSEQ, n = row % NSEQ;
  const int tid = threadIdx.x;

  float vals[4];
  float s = 0.0f;
#pragma unroll
  for (int i = 0; i < 4; ++i) {
    int c = tid + i * 256;
    int h = c >> 6, d = c & 63;
    vals[i] = attn_out[(((size_t)(b * NHEADS + h)) * NSEQ + n) * HD + d];
    s += vals[i];
  }
  red[tid] = s; __syncthreads();
  for (int off = 128; off > 0; off >>= 1) {
    if (tid < off) red[tid] += red[tid + off];
    __syncthreads();
  }
  const float mu = red[0] * (1.0f / DIMC);
  __syncthreads();

  float sq = 0.0f;
#pragma unroll
  for (int i = 0; i < 4; ++i) { float d = vals[i] - mu; sq += d * d; }
  red[tid] = sq; __syncthreads();
  for (int off = 128; off > 0; off >>= 1) {
    if (tid < off) red[tid] += red[tid + off];
    __syncthreads();
  }
  const float var = red[0] * (1.0f / DIMC);
  const float inv = rsqrtf(var + 1e-5f);

#pragma unroll
  for (int i = 0; i < 4; ++i) {
    int c = tid + i * 256;
    yh[(size_t)row * DIMC + c] = (_Float16)((vals[i] - mu) * inv * gamma[c] + beta[c]);
  }
}

// ---------------------------------------------------------------- launch
extern "C" void kernel_launch(void* const* d_in, const int* in_sizes, int n_in,
                              void* d_out, int out_size, void* d_ws, size_t ws_size,
                              hipStream_t stream) {
  const float* x      = (const float*)d_in[0];
  const float* pe     = (const float*)d_in[1];
  const float* qkv_w  = (const float*)d_in[2];
  const float* q_bias = (const float*)d_in[3];
  const float* v_bias = (const float*)d_in[4];
  const float* gamma  = (const float*)d_in[5];
  const float* beta   = (const float*)d_in[6];
  const float* proj_w = (const float*)d_in[7];
  const float* proj_b = (const float*)d_in[8];

  float* outp  = (float*)d_out;                         // [B,N,1024]
  float* kmean = outp + (size_t)MROWS * DIMC;           // [B,N,64]

  char* ws = (char*)d_ws;
  _Float16* xh   = (_Float16*)(ws);                     //  0.. 8 MB (reused as yh)
  _Float16* wh   = (_Float16*)(ws + (size_t)8  * 1048576);  //  8..14 MB
  _Float16* pwh  = (_Float16*)(ws + (size_t)14 * 1048576);  // 14..16 MB
  float*    qkv  = (float*)   (ws + (size_t)16 * 1048576);  // 16..64 MB (reused as attn_out)
  _Float16* qh   = (_Float16*)(ws + (size_t)64 * 1048576);  // 64..72 MB
  _Float16* kh   = (_Float16*)(ws + (size_t)72 * 1048576);  // 72..80 MB
  _Float16* vt   = (_Float16*)(ws + (size_t)80 * 1048576);  // 80..88 MB
  float*    atn  = qkv;                                 // reuse (qkv dead after rope)
  _Float16* yh   = xh;                                  // reuse (xh dead after gemm1)

  // 1) f32 -> f16 conversions
  f32_to_f16_kernel<<<(MROWS * DIMC + 255) / 256, 256, 0, stream>>>(x, xh, MROWS * DIMC);
  f32_to_f16_kernel<<<(QKV_N * DIMC + 255) / 256, 256, 0, stream>>>(qkv_w, wh, QKV_N * DIMC);
  f32_to_f16_kernel<<<(DIMC * DIMC + 255) / 256, 256, 0, stream>>>(proj_w, pwh, DIMC * DIMC);

  // 2) QKV GEMM (bias folded later in rope pass)
  gemm_wmma_kernel<<<dim3(QKV_N / 256, MROWS / 32), 128, 0, stream>>>(
      xh, wh, nullptr, qkv, MROWS, QKV_N, DIMC);

  // 3) RoPE + bias + pack (q scaled, v transposed)
  rope_pack_kernel<<<(BQ * NHEADS * NSEQ * 32) / 256, 256, 0, stream>>>(
      qkv, pe, q_bias, v_bias, qh, kh, vt);

  // 4) k_mean output
  kmean_kernel<<<(BQ * NSEQ * HD) / 256, 256, 0, stream>>>(kh, kmean);

  // 5) attention (flash-style, WMMA)
  attn_kernel<<<dim3(NSEQ / 64, NHEADS, BQ), 128, 0, stream>>>(qh, kh, vt, atn);

  // 6) layernorm -> f16
  ln_kernel<<<MROWS, 256, 0, stream>>>(atn, gamma, beta, yh);

  // 7) projection GEMM + bias -> d_out
  gemm_wmma_kernel<<<dim3(DIMC / 256, MROWS / 32), 128, 0, stream>>>(
      yh, pwh, proj_b, outp, MROWS, DIMC, DIMC);
}
